// TongDiGCN_50818053046715
// MI455X (gfx1250) — matrix operand order repaired and checked by reading, not compiled, not run on hardware
//
#include <hip/hip_runtime.h>

typedef _Float16 half_t;
typedef _Float16 v16h __attribute__((ext_vector_type(16)));
typedef _Float16 h2v  __attribute__((ext_vector_type(2)));
typedef _Float16 h4v  __attribute__((ext_vector_type(4)));
typedef float    v4f  __attribute__((ext_vector_type(4)));
typedef float    v8f  __attribute__((ext_vector_type(8)));

#define DH 256   // hidden width (also input width); hardcoded per reference

// ---------------- degree / normalization ----------------

__global__ void deg_init(float* __restrict__ degf, float* __restrict__ degb, int n) {
  int i = blockIdx.x * blockDim.x + threadIdx.x;
  if (i < n) { degf[i] = 1.0f; degb[i] = 1.0f; }   // self-loop contributes 1
}

__global__ void deg_count(const int* __restrict__ src, const int* __restrict__ dst,
                          float* __restrict__ degf, float* __restrict__ degb, int E) {
  int e = blockIdx.x * blockDim.x + threadIdx.x;
  if (e < E) {
    atomicAdd(&degf[dst[e]], 1.0f);  // forward: in-degree at dst
    atomicAdd(&degb[src[e]], 1.0f);  // backward (edges reversed): degree at src
  }
}

__global__ void deg_to_dinv(float* __restrict__ degf, float* __restrict__ degb, int n) {
  int i = blockIdx.x * blockDim.x + threadIdx.x;
  if (i < n) { degf[i] = rsqrtf(degf[i]); degb[i] = rsqrtf(degb[i]); }
}

__global__ void zero_buf(float* __restrict__ p, long long count) {
  long long i = (long long)blockIdx.x * blockDim.x + threadIdx.x;
  if (i < count) p[i] = 0.0f;
}

// ---------------- WMMA GEMM: C = A[MxK] @ W[KxN] (+bias) (+=C) ----------------
// Split-f16 precision: A=Ah+Al, B=Bh+Bl; acc += Ah*Bh + Ah*Bl + Al*Bh (f32 accum).
// Block tile 64(M) x 32(N), 8 waves (4x2 of 16x16 WMMA tiles), K-step 32.
// Branchless, vectorized (B128) global loads; flags are template params so the
// writeback epilogue is straight-line code in every instantiation.

#define LDSP 36  // padded row stride (halves); 36*2=72B keeps h4 stores 8B-aligned

template <bool HAS_BIAS, bool ACC>
__global__ __launch_bounds__(256)
void gemm_wmma(const float* __restrict__ A, const float* __restrict__ W,
               float* __restrict__ C, const float* __restrict__ bias,
               int M, int N, int K) {
  __shared__ half_t AsH[64][LDSP];
  __shared__ half_t AsL[64][LDSP];
  __shared__ half_t BsH[32][LDSP];   // stored transposed: Bs[n][k]
  __shared__ half_t BsL[32][LDSP];

  const int tid  = threadIdx.x;
  const int lane = tid & 31;
  const int wave = tid >> 5;
  const int wm   = wave & 3;          // M sub-tile 0..3
  const int wn   = wave >> 2;         // N sub-tile 0..1
  const int m0   = blockIdx.x * 64;
  const int n0   = blockIdx.y * 32;

  const int mrow = wm * 16 + (lane & 15);
  const int ncol = wn * 16 + (lane & 15);
  const int h    = lane >> 4;

  // per-thread tile-load coordinates (float4 granularity)
  const int la_r = tid >> 3;          // 0..31 (A rows, +32 on 2nd iter)
  const int la_c = (tid & 7) * 4;     // 0..28 (cols, 16B aligned)

  v8f acc = {};
  for (int k0 = 0; k0 < K; k0 += 32) {
    // stage A tile 64x32: two float4 loads per thread, branchless edge handling
#pragma unroll
    for (int i = 0; i < 2; ++i) {
      int r   = la_r + i * 32;
      int gm  = m0 + r;
      int gmc = (gm < M) ? gm : (M - 1);
      v4f v = *(const v4f*)&A[(long long)gmc * K + k0 + la_c];
      float keep = (gm < M) ? 1.0f : 0.0f;
      h4v hi, lo;
#pragma unroll
      for (int j = 0; j < 4; ++j) {
        float f = v[j] * keep;
        half_t hj = (half_t)f;
        hi[j] = hj;
        lo[j] = (half_t)(f - (float)hj);
      }
      *(h4v*)&AsH[r][la_c] = hi;      // single ds_store_b64 each
      *(h4v*)&AsL[r][la_c] = lo;
    }
    // stage B tile 32x32 transposed: one float4 load per thread
    {
      int kk = la_r;                  // 0..31 (K row)
      v4f v = *(const v4f*)&W[(long long)(k0 + kk) * N + n0 + la_c];
#pragma unroll
      for (int j = 0; j < 4; ++j) {
        half_t hj = (half_t)v[j];
        BsH[la_c + j][kk] = hj;
        BsL[la_c + j][kk] = (half_t)(v[j] - (float)hj);
      }
    }
    // prefetch next K-step tiles into cache while we compute on this one
    if (k0 + 32 < K) {
      int gmn = m0 + la_r; if (gmn >= M) gmn = M - 1;
      __builtin_prefetch(&A[(long long)gmn * K + k0 + 32 + la_c], 0, 3);
      __builtin_prefetch(&W[(long long)(k0 + 32 + la_r) * N + n0 + la_c], 0, 3);
    }
    __syncthreads();

    // build fragments per ISA 16-bit layouts (wave32)
    v16h ah, al, bh, bl;
#pragma unroll
    for (int v = 0; v < 8; ++v) {
      int ka = ((v < 4) ? 0 : 16) + h * 8 + (v & 3) * 2;   // A: 16x32 layout
      h2v pah = *(const h2v*)&AsH[mrow][ka];
      h2v pal = *(const h2v*)&AsL[mrow][ka];
      ah[2 * v] = pah.x; ah[2 * v + 1] = pah.y;
      al[2 * v] = pal.x; al[2 * v + 1] = pal.y;
      int kb = ((v < 4) ? 0 : 16) + h * 8 + (v & 3) * 2;   // B: 32x16 layout
      h2v pbh = *(const h2v*)&BsH[ncol][kb];
      h2v pbl = *(const h2v*)&BsL[ncol][kb];
      bh[2 * v] = pbh.x; bh[2 * v + 1] = pbh.y;
      bl[2 * v] = pbl.x; bl[2 * v + 1] = pbl.y;
    }
    acc = __builtin_amdgcn_wmma_f32_16x16x32_f16(false, ah, false, bh, (short)0, acc, false, false);
    acc = __builtin_amdgcn_wmma_f32_16x16x32_f16(false, ah, false, bl, (short)0, acc, false, false);
    acc = __builtin_amdgcn_wmma_f32_16x16x32_f16(false, al, false, bh, (short)0, acc, false, false);
    __syncthreads();
  }

  // writeback: VGPR r -> M = r + (lane/16)*8, N = lane%16; straight-line epilogue
  const int gn    = n0 + wn * 16 + (lane & 15);
  const int mbase = m0 + wm * 16 + (lane >> 4) * 8;
  const float bv  = HAS_BIAS ? bias[gn] : 0.0f;
  if (m0 + 64 <= M) {                 // uniform fast path: no per-lane guards
    float* __restrict__ cp = C + (long long)mbase * N + gn;
#pragma unroll
    for (int r = 0; r < 8; ++r) {
      float v = acc[r] + bv;
      if (ACC) v += cp[(long long)r * N];
      cp[(long long)r * N] = v;
    }
  } else {
#pragma unroll
    for (int r = 0; r < 8; ++r) {
      int gm = mbase + r;
      if (gm < M) {
        long long o = (long long)gm * N + gn;
        float v = acc[r] + bv;
        if (ACC) v += C[o];
        C[o] = v;
      }
    }
  }
}

// ---------------- edge aggregation ----------------
// accum[d] += h[s] * dinv[s] * dinv[d]; one block (256 lanes = DH feats) per edge.

__global__ __launch_bounds__(256)
void edge_scatter(const float* __restrict__ h, const float* __restrict__ dinv,
                  const int* __restrict__ s_idx, const int* __restrict__ d_idx,
                  float* __restrict__ accum) {
  int e = blockIdx.x;
  int s = s_idx[e], d = d_idx[e];
  float norm = dinv[s] * dinv[d];
  int f = threadIdx.x;
  atomicAdd(&accum[(long long)d * DH + f], h[(long long)s * DH + f] * norm);
}

// self-loop term + bias (+ ReLU), in place on the accumulation buffer
__global__ __launch_bounds__(256)
void node_epilogue(const float* __restrict__ h, const float* __restrict__ dinv,
                   const float* __restrict__ bias, float* __restrict__ accum,
                   int relu) {
  int i = blockIdx.x, f = threadIdx.x;
  float di = dinv[i];
  long long o = (long long)i * DH + f;
  float v = accum[o] + h[o] * di * di + bias[f];
  if (relu) v = fmaxf(v, 0.0f);
  accum[o] = v;
}

// ---------------- driver ----------------

extern "C" void kernel_launch(void* const* d_in, const int* in_sizes, int n_in,
                              void* d_out, int out_size, void* d_ws, size_t ws_size,
                              hipStream_t stream) {
  const float* x    = (const float*)d_in[0];
  const int*   ei   = (const int*)d_in[1];
  const float* Wf1  = (const float*)d_in[2];
  const float* bf1  = (const float*)d_in[3];
  const float* Wf2  = (const float*)d_in[4];
  const float* bf2  = (const float*)d_in[5];
  const float* Wb1  = (const float*)d_in[6];
  const float* bb1  = (const float*)d_in[7];
  const float* Wb2  = (const float*)d_in[8];
  const float* bb2  = (const float*)d_in[9];
  const float* Wfin = (const float*)d_in[10];
  const float* bfin = (const float*)d_in[11];
  float* out = (float*)d_out;

  const int n = in_sizes[0] / DH;
  const int E = in_sizes[1] / 2;
  const int* src = ei;
  const int* dst = ei + E;

  float* ws    = (float*)d_ws;
  float* dinvf = ws;
  float* dinvb = ws + n;
  float* bufA  = ws + 2 * (long long)n;              // GEMM outputs (h)
  float* bufB  = bufA + (long long)n * DH;           // aggregation / layer activations

  const long long nd = (long long)n * DH;
  const int zblk = (int)((nd + 255) / 256);

  // normalization vectors (shared by both layers of each direction)
  deg_init   <<<(n + 255) / 256, 256, 0, stream>>>(dinvf, dinvb, n);
  deg_count  <<<(E + 255) / 256, 256, 0, stream>>>(src, dst, dinvf, dinvb, E);
  deg_to_dinv<<<(n + 255) / 256, 256, 0, stream>>>(dinvf, dinvb, n);

  dim3 blk(256);
  dim3 g256((n + 63) / 64, DH / 32);
  dim3 g128((n + 63) / 64, 128 / 32);

  // ---- forward direction (s=src, d=dst) ----
  gemm_wmma<false, false><<<g256, blk, 0, stream>>>(x, Wf1, bufA, nullptr, n, DH, DH);
  zero_buf<<<zblk, 256, 0, stream>>>(bufB, nd);
  edge_scatter<<<E, 256, 0, stream>>>(bufA, dinvf, src, dst, bufB);
  node_epilogue<<<n, 256, 0, stream>>>(bufA, dinvf, bf1, bufB, 1);
  gemm_wmma<false, false><<<g256, blk, 0, stream>>>(bufB, Wf2, bufA, nullptr, n, DH, DH);
  zero_buf<<<zblk, 256, 0, stream>>>(bufB, nd);
  edge_scatter<<<E, 256, 0, stream>>>(bufA, dinvf, src, dst, bufB);
  node_epilogue<<<n, 256, 0, stream>>>(bufA, dinvf, bf2, bufB, 0);
  // out = x_fwd @ Wfin[0:256,:] + bfin
  gemm_wmma<true, false><<<g128, blk, 0, stream>>>(bufB, Wfin, out, bfin, n, 128, DH);

  // ---- backward direction (edges reversed: s=dst, d=src) ----
  gemm_wmma<false, false><<<g256, blk, 0, stream>>>(x, Wb1, bufA, nullptr, n, DH, DH);
  zero_buf<<<zblk, 256, 0, stream>>>(bufB, nd);
  edge_scatter<<<E, 256, 0, stream>>>(bufA, dinvb, dst, src, bufB);
  node_epilogue<<<n, 256, 0, stream>>>(bufA, dinvb, bb1, bufB, 1);
  gemm_wmma<false, false><<<g256, blk, 0, stream>>>(bufB, Wb2, bufA, nullptr, n, DH, DH);
  zero_buf<<<zblk, 256, 0, stream>>>(bufB, nd);
  edge_scatter<<<E, 256, 0, stream>>>(bufA, dinvb, dst, src, bufB);
  node_epilogue<<<n, 256, 0, stream>>>(bufA, dinvb, bb2, bufB, 0);
  // out += x_bwd @ Wfin[256:512,:]
  gemm_wmma<false, true><<<g128, blk, 0, stream>>>(bufB, Wfin + 256 * 128, out, nullptr, n, 128, DH);
}
